// SwitchFocusedLoss_80221399155360
// MI455X (gfx1250) — compile-verified, matrix operand order checked by reading, not run on hardware
//
#include <hip/hip_runtime.h>
#include <hip/hip_bf16.h>

// Problem dims (fixed by the reference)
#define B_DIM 32
#define S_DIM 8192
#define C_DIM 128
#define N_POS (B_DIM * S_DIM)   // 262144 positions

typedef __attribute__((ext_vector_type(2)))  float    v2f;
typedef __attribute__((ext_vector_type(4)))  float    v4f;
typedef __attribute__((ext_vector_type(8)))  float    v8f;
typedef __attribute__((ext_vector_type(16))) _Float16 v16h;

// ---------------------------------------------------------------------------
// Kernel 1: per-position CE + argmax + flags. One wave32 per position.
// Lane l owns classes [4l, 4l+3] -> one 16B coalesced load per lane
// (512 contiguous bytes per wave). Non-temporal: logits are read once
// (134 MB) and should not churn L2.
// ---------------------------------------------------------------------------
__global__ __launch_bounds__(256) void ce_kernel(
    const float* __restrict__ logits,
    const int*   __restrict__ labels,
    const int*   __restrict__ seg_marks,
    float*        __restrict__ ce_out,
    unsigned char* __restrict__ flags_out) {
  const int wave = threadIdx.x >> 5;
  const int lane = threadIdx.x & 31;
  const int pos  = blockIdx.x * 8 + wave;          // always < N_POS

  const v4f* row = (const v4f*)(logits + (size_t)pos * C_DIM) + lane;
  v4f v = __builtin_nontemporal_load(row);

  // Speculative prefetch of the row this wave handles 8 positions later
  // (gfx1250: global_prefetch_b8; translation failures silently dropped).
  __builtin_prefetch((const char*)(row) + 8 * C_DIM * sizeof(float), 0, 0);

  // Local max/argmax over 4 elements (first-max tie-break like jnp.argmax).
  float m  = v.x; int mi = 0;
  if (v.y > m) { m = v.y; mi = 1; }
  if (v.z > m) { m = v.z; mi = 2; }
  if (v.w > m) { m = v.w; mi = 3; }
  float amax = m;
  int   aidx = lane * 4 + mi;

  // Wave32 butterfly reduce: lexicographic max on (value, -index).
#pragma unroll
  for (int off = 16; off > 0; off >>= 1) {
    float om = __shfl_xor(amax, off, 32);
    int   oi = __shfl_xor(aidx, off, 32);
    if (om > amax || (om == amax && oi < aidx)) { amax = om; aidx = oi; }
  }
  const float rowmax = amax;                       // identical in all lanes

  // Sum of exp(x - max) over 128 classes.
  float e = __expf(v.x - rowmax) + __expf(v.y - rowmax) +
            __expf(v.z - rowmax) + __expf(v.w - rowmax);
#pragma unroll
  for (int off = 16; off > 0; off >>= 1) e += __shfl_xor(e, off, 32);

  const int  lab   = labels[pos];                  // uniform across the wave
  const bool valid = (lab != -100);
  const int  slab  = valid ? lab : 0;

  // logit[slab]: slab is wave-uniform, so select component then broadcast.
  float cand;
  switch (slab & 3) {
    case 0:  cand = v.x; break;
    case 1:  cand = v.y; break;
    case 2:  cand = v.z; break;
    default: cand = v.w; break;
  }
  const float tgt = __shfl(cand, slab >> 2, 32);
  const float ce  = rowmax + __logf(e) - tgt;      // -log_softmax[label]

  if (lane == 0) {
    ce_out[pos] = valid ? ce : 0.0f;
    const bool pred_sw = (aidx == 2) || (aidx == 3);
    const bool is_sw   = (lab == 2) || (lab == 3);
    const bool sg      = seg_marks[pos] > 0;
    flags_out[pos] = (unsigned char)((valid ? 1u : 0u) | (is_sw ? 2u : 0u) |
                                     (pred_sw ? 4u : 0u) | (sg ? 8u : 0u));
  }
}

// ---------------------------------------------------------------------------
// Exact f32 reduction of 256 LDS floats by a single wave via WMMA.
// D = A x B with B == ones  =>  D[i][j] = sum_k A[i][k]  (row sums).
// 4 chained V_WMMA_F32_16X16X4_F32 cover 16 rows x 16 K = 256 values,
// each value mapped exactly once; then 8 VGPR adds + shfl_xor(16).
// Must be executed with all 32 lanes of the wave active (EXEC == ~0).
// ---------------------------------------------------------------------------
#if __has_builtin(__builtin_amdgcn_wmma_f32_16x16x4_f32)
__device__ __forceinline__ float wmma_reduce256(const float* sm, int lane) {
  const int M  = lane & 15;          // A-matrix row for this lane
  const int kb = (lane >> 4) * 2;    // lanes 0-15: K={0,1}; lanes 16-31: K={2,3}
  v2f b; b.x = 1.0f; b.y = 1.0f;     // ones B-matrix (layout-independent)
  v8f c = {};
#pragma unroll
  for (int ch = 0; ch < 4; ++ch) {   // K chunks of 4 -> 16 values per row
    v2f a;
    a.x = sm[M * 16 + ch * 4 + kb + 0];
    a.y = sm[M * 16 + ch * 4 + kb + 1];
    c = __builtin_amdgcn_wmma_f32_16x16x4_f32(
        /*neg_a=*/false, a, /*neg_b=*/false, b,
        /*c_mod=*/(short)0, c, /*reuse_a=*/false, /*reuse_b=*/false);
  }
  // Lane j holds rows 0-7 (lanes 0-15) or rows 8-15 (lanes 16-31), column j.
  float r = c[0] + c[1] + c[2] + c[3] + c[4] + c[5] + c[6] + c[7];
  r += __shfl_xor(r, 16, 32);        // merge the two row halves
  return r;
}
#else
// Fallback: codegen-confirmed f16 WMMA; values pre-scaled to stay in f16 range.
__device__ __forceinline__ float wmma_reduce256(const float* sm, int lane) {
  const int M    = lane & 15;
  const int base = M * 16 + (lane >> 4) * 8;   // 8 consecutive values per lane
  const float sc = 1.0f / 1024.0f;
  v16h a;
#pragma unroll
  for (int i = 0; i < 8; ++i)  a[i] = (_Float16)(sm[base + i] * sc);
#pragma unroll
  for (int i = 8; i < 16; ++i) a[i] = (_Float16)0.0f;   // zero-pad K=16..31
  v16h bo;
#pragma unroll
  for (int i = 0; i < 16; ++i) bo[i] = (_Float16)1.0f;
  v8f c = {};
  c = __builtin_amdgcn_wmma_f32_16x16x32_f16(false, a, false, bo,
                                             (short)0, c, false, false);
  float r = c[0] + c[1] + c[2] + c[3] + c[4] + c[5] + c[6] + c[7];
  r += __shfl_xor(r, 16, 32);
  return r * 1024.0f;
}
#endif

// ---------------------------------------------------------------------------
// Kernel 2: neighborhood weights (+-5 along S, row-local) + block reduction.
// Flags/CE arrays total ~1.3 MB -> L2 resident on a 192 MB L2.
// ---------------------------------------------------------------------------
__global__ __launch_bounds__(256) void weight_reduce_kernel(
    const float*         __restrict__ ce_in,
    const unsigned char* __restrict__ flags,
    float*               __restrict__ acc) {   // acc[0]=sum(ce*w*vf), acc[1]=sum(vf)
  __shared__ float s_num[256];
  __shared__ float s_den[256];

  const int p = blockIdx.x * 256 + threadIdx.x;  // < N_POS
  const int s = p & (S_DIM - 1);                 // position within its row

  const unsigned char f = flags[p];
  const bool valid   = f & 1;
  const bool is_sw   = f & 2;
  const bool pred_sw = f & 4;
  const bool sg      = f & 8;

  // log(2 - d/6) for d = 1..5
  const float LF[5] = {0.60613580f, 0.51082562f, 0.40546511f,
                       0.28768207f, 0.15415068f};
  float log_left = 0.0f, log_right = 0.0f;
#pragma unroll
  for (int d = 1; d <= 5; ++d) {
    if (s - d >= 0     && (flags[p - d] & 2)) log_left  += LF[d - 1];
    if (s + d <  S_DIM && (flags[p + d] & 2)) log_right += LF[d - 1];
  }

  float w;
  if (is_sw) {
    const float base = pred_sw ? 10.0f : 20.0f;
    w = pred_sw ? base * __expf(log_right) : base;
  } else {
    w = (pred_sw && valid) ? __expf(log_left + log_right) : 1.0f;
  }
  if (sg && valid && !is_sw) w *= pred_sw ? 3.0f : 0.3f;

  s_num[threadIdx.x] = ce_in[p] * w;             // ce_in is 0 where invalid
  s_den[threadIdx.x] = valid ? 1.0f : 0.0f;
  __syncthreads();

  if (threadIdx.x < 32) {                        // wave 0, all 32 lanes active
    const int lane = threadIdx.x;
    const float tn = wmma_reduce256(s_num, lane);
    const float td = wmma_reduce256(s_den, lane);
    if (lane == 0) {
      atomicAdd(&acc[0], tn);
      atomicAdd(&acc[1], td);
    }
  }
}

__global__ void init_acc(float* acc) {
  if (threadIdx.x < 2) acc[threadIdx.x] = 0.0f;
}

__global__ void finalize(const float* __restrict__ acc, float* __restrict__ out) {
  out[0] = acc[0] / acc[1];
}

// ---------------------------------------------------------------------------
// Launch. Workspace layout: [ce: N floats][flags: N bytes][acc: 2 floats]
// = 1,310,728 bytes total.
// ---------------------------------------------------------------------------
extern "C" void kernel_launch(void* const* d_in, const int* in_sizes, int n_in,
                              void* d_out, int out_size, void* d_ws, size_t ws_size,
                              hipStream_t stream) {
  (void)in_sizes; (void)n_in; (void)out_size; (void)ws_size;

  const float* logits    = (const float*)d_in[0];
  const int*   labels    = (const int*)d_in[1];
  const int*   seg_marks = (const int*)d_in[2];

  float*         ce    = (float*)d_ws;
  unsigned char* flags = (unsigned char*)d_ws + (size_t)N_POS * sizeof(float);
  float*         acc   = (float*)((char*)d_ws + (size_t)N_POS * 5);

  init_acc<<<1, 32, 0, stream>>>(acc);
  ce_kernel<<<N_POS / 8, 256, 0, stream>>>(logits, labels, seg_marks, ce, flags);
  weight_reduce_kernel<<<N_POS / 256, 256, 0, stream>>>(ce, flags, acc);
  finalize<<<1, 1, 0, stream>>>(acc, (float*)d_out);
}
